// QwenMoeWrapperSkipBaseline_44418551775978
// MI455X (gfx1250) — matrix operand order repaired and checked by reading, not compiled, not run on hardware
//
#include <hip/hip_runtime.h>
#include <hip/hip_bf16.h>
#include <stdint.h>

// Problem constants (from reference): B=2, S=512 -> N=1024 tokens
#define NTOK 1024
#define HDIM 2048
#define NEXP 16
#define IDIM 1408

// Block tile
#define BM 128
#define BN 64
#define BK 32

typedef __bf16 bf16_t;
typedef __attribute__((ext_vector_type(16))) __bf16 v16bf;
typedef __attribute__((ext_vector_type(8)))  float  v8f;

union Frag16 { v16bf v; uint32_t u[8]; };
union FragC  { v8f   v; float    f[8]; };

__device__ __forceinline__ bf16_t to_bf16(float x) { return (bf16_t)x; }

// ---------------------------------------------------------------------------
// Kernel 1: router. One block (256 thr) per token.
// ---------------------------------------------------------------------------
__global__ __launch_bounds__(256)
void router_kernel(const float* __restrict__ x,
                   const float* __restrict__ gate_w,
                   bf16_t* __restrict__ xb,
                   int*   __restrict__ counts,
                   int*   __restrict__ etok,
                   float* __restrict__ ewt) {
  const int row = blockIdx.x;
  const int t   = threadIdx.x;
  const float* xr = x + (size_t)row * HDIM;

  for (int i = t; i < HDIM; i += 256)
    xb[(size_t)row * HDIM + i] = to_bf16(xr[i]);

  __shared__ float part[NEXP][17];
  __shared__ float logits[NEXP];

  const int e = t & 15;
  const int g = t >> 4;
  const float* we = gate_w + (size_t)e * HDIM;
  float acc = 0.f;
  for (int i = g; i < HDIM; i += 16)
    acc += xr[i] * we[i];
  part[e][g] = acc;
  __syncthreads();

  if (t < NEXP) {
    float s = 0.f;
    #pragma unroll
    for (int i = 0; i < 16; ++i) s += part[t][i];
    logits[t] = s;
  }
  __syncthreads();

  if (t == 0) {
    float l0 = -3.4e38f, l1 = -3.4e38f;
    int   i0 = 0, i1 = 0;
    #pragma unroll
    for (int k = 0; k < NEXP; ++k) {
      float v = logits[k];
      if (v > l0)      { l1 = l0; i1 = i0; l0 = v; i0 = k; }
      else if (v > l1) { l1 = v;  i1 = k; }
    }
    // top-4 selected, first keep_k=2 kept, renormalized -> softmax over top-2
    float p1 = __expf(l1 - l0);
    float s  = 1.f + p1;
    float w0 = 1.f / s;
    float w1 = p1  / s;
    int pos0 = atomicAdd(&counts[i0], 1);
    etok[i0 * NTOK + pos0] = row;  ewt[i0 * NTOK + pos0] = w0;
    int pos1 = atomicAdd(&counts[i1], 1);
    etok[i1 * NTOK + pos1] = row;  ewt[i1 * NTOK + pos1] = w1;
  }
}

// ---------------------------------------------------------------------------
// Fragment helpers (ISA 16-bit A 16x32 / B 32x16 VGPR layouts)
// ---------------------------------------------------------------------------
__device__ __forceinline__ void load_a_frag(Frag16& af, const bf16_t* rowp,
                                            int hi) {
  const uint32_t* arow = (const uint32_t*)rowp;
  #pragma unroll
  for (int v = 0; v < 8; ++v) {
    int k = ((v & 4) << 2) + hi * 8 + (v & 3) * 2;
    af.u[v] = arow[k >> 1];
  }
}
__device__ __forceinline__ void load_b_frag(Frag16& bf_, const bf16_t* rowp,
                                            int hi) {
  const uint32_t* brow = (const uint32_t*)rowp;
  #pragma unroll
  for (int v = 0; v < 8; ++v)
    bf_.u[v] = brow[(((v & 4) << 2) + hi * 8 + (v & 3) * 2) >> 1];
}

// ---------------------------------------------------------------------------
// Kernel 2: grouped gate/up GEMM + fused SwiGLU.
// Block tile 128(M) x 64(N), 8 waves as 4x2, each wave 32x32 per matrix.
// ---------------------------------------------------------------------------
__global__ __launch_bounds__(256)
void gemm1_kernel(const bf16_t* __restrict__ xb,
                  const float*  __restrict__ w_gate,
                  const float*  __restrict__ w_up,
                  const int*    __restrict__ counts,
                  const int*    __restrict__ etok,
                  bf16_t*       __restrict__ act) {
  const int e  = blockIdx.z;
  const int m0 = blockIdx.y * BM;
  const int n0 = blockIdx.x * BN;
  const int cnt = counts[e];
  if (m0 >= cnt) return;

  const int t = threadIdx.x;
  __shared__ bf16_t As[BM][34];
  __shared__ bf16_t Bg[BN][34];
  __shared__ bf16_t Bu[BN][34];
  __shared__ int    tokS[BM];

  if (t < BM) {
    int slot = m0 + t;
    tokS[t] = (slot < cnt) ? etok[e * NTOK + slot] : 0;
  }
  if (t >= BM && t < BM + 64) {
  }
  // (BM==128 > 64 lanes? no: 256 threads cover BM=128 directly above)
  __syncthreads();

  const int lane = t & 31;
  const int lr   = lane & 15;
  const int hi   = lane >> 4;
  const int wid  = t >> 5;
  const int mB   = (wid & 3) * 32;   // wave M offset (0/32/64/96)
  const int wn   = wid >> 2;         // wave N half (0/1)

  const float* wg = w_gate + (size_t)e * HDIM * IDIM;
  const float* wu = w_up   + (size_t)e * HDIM * IDIM;

  // Hoisted staging addresses ------------------------------------------------
  // A: pair index p = t + 256*i -> m = (t>>4) + 16*i, kp = t&15
  const int amb = t >> 4;          // 0..15
  const int akp = t & 15;          // pair column
  const uint32_t* aRow[8];
  #pragma unroll
  for (int i = 0; i < 8; ++i)
    aRow[i] = (const uint32_t*)(xb + (size_t)tokS[amb + 16 * i] * HDIM) + akp;
  // B: idx = t + 256*i -> k = (t>>6) + 4*i, n = t&63
  const int bkb = t >> 6;          // 0..3
  const int bnn = t & 63;
  const float* pg0 = wg + (size_t)bkb * IDIM + n0 + bnn;
  const float* pu0 = wu + (size_t)bkb * IDIM + n0 + bnn;

  FragC accg[2][2], accu[2][2];
  #pragma unroll
  for (int a = 0; a < 2; ++a)
    #pragma unroll
    for (int j = 0; j < 2; ++j)
      #pragma unroll
      for (int r = 0; r < 8; ++r) { accg[a][j].f[r] = 0.f; accu[a][j].f[r] = 0.f; }

  for (int k0 = 0; k0 < HDIM; k0 += BK) {
    __syncthreads();
    // Stage A: 128 rows x 32 K (bf16 pairs, dword moves)
    #pragma unroll
    for (int i = 0; i < 8; ++i)
      *(uint32_t*)&As[amb + 16 * i][akp * 2] = aRow[i][k0 >> 1];
    // Stage B: 32 K x 64 N of w_gate/w_up, fp32 -> bf16, transposed [n][k]
    const float* pg = pg0 + (size_t)k0 * IDIM;
    const float* pu = pu0 + (size_t)k0 * IDIM;
    #pragma unroll
    for (int i = 0; i < 8; ++i) {
      Bg[bnn][bkb + 4 * i] = to_bf16(pg[(size_t)(4 * i) * IDIM]);
      Bu[bnn][bkb + 4 * i] = to_bf16(pu[(size_t)(4 * i) * IDIM]);
    }
    if (k0 + BK < HDIM) {
      __builtin_prefetch(pg + (size_t)BK * IDIM, 0, 0);
      __builtin_prefetch(pu + (size_t)BK * IDIM, 0, 0);
    }
    __syncthreads();

    Frag16 af[2], bg[2], bu[2];
    load_a_frag(af[0], &As[mB + lr][0], hi);
    load_a_frag(af[1], &As[mB + 16 + lr][0], hi);
    #pragma unroll
    for (int j = 0; j < 2; ++j) {
      const int nb = wn * 32 + j * 16 + lr;
      load_b_frag(bg[j], &Bg[nb][0], hi);
      load_b_frag(bu[j], &Bu[nb][0], hi);
    }
    #pragma unroll
    for (int a = 0; a < 2; ++a)
      #pragma unroll
      for (int j = 0; j < 2; ++j) {
        accg[a][j].v = __builtin_amdgcn_wmma_f32_16x16x32_bf16(
            false, af[a].v, false, bg[j].v, (short)0, accg[a][j].v, false, false);
        accu[a][j].v = __builtin_amdgcn_wmma_f32_16x16x32_bf16(
            false, af[a].v, false, bu[j].v, (short)0, accu[a][j].v, false, false);
      }
  }

  // Fused SwiGLU epilogue -> bf16 activations at [e][slot][I]
  bf16_t* actE = act + (size_t)e * NTOK * IDIM;
  #pragma unroll
  for (int a = 0; a < 2; ++a)
    #pragma unroll
    for (int j = 0; j < 2; ++j) {
      const int n = n0 + wn * 32 + j * 16 + lr;
      #pragma unroll
      for (int r = 0; r < 8; ++r) {
        const int slot = m0 + mB + a * 16 + hi * 8 + r;
        float g = accg[a][j].f[r];
        float u = accu[a][j].f[r];
        float v = g * (1.0f / (1.0f + __expf(-g))) * u;
        actE[(size_t)slot * IDIM + n] = to_bf16(v);
      }
    }
}

// ---------------------------------------------------------------------------
// Kernel 3: grouped down GEMM, weighted atomic scatter-add into fp32 output.
// ---------------------------------------------------------------------------
__global__ __launch_bounds__(256)
void gemm2_kernel(const bf16_t* __restrict__ act,
                  const float*  __restrict__ w_down,
                  const int*    __restrict__ counts,
                  const int*    __restrict__ etok,
                  const float*  __restrict__ ewt,
                  float*        __restrict__ out) {
  const int e  = blockIdx.z;
  const int m0 = blockIdx.y * BM;
  const int n0 = blockIdx.x * BN;
  const int cnt = counts[e];
  if (m0 >= cnt) return;

  const int t = threadIdx.x;
  __shared__ bf16_t As[BM][34];
  __shared__ bf16_t Bd[BN][34];
  __shared__ int    tokS[BM];
  __shared__ float  wtS[BM];

  if (t < BM) {
    int slot = m0 + t;
    tokS[t] = (slot < cnt) ? etok[e * NTOK + slot] : 0;
    wtS[t]  = (slot < cnt) ? ewt[e * NTOK + slot] : 0.f;
  }
  __syncthreads();

  const int lane = t & 31;
  const int lr   = lane & 15;
  const int hi   = lane >> 4;
  const int wid  = t >> 5;
  const int mB   = (wid & 3) * 32;
  const int wn   = wid >> 2;

  const bf16_t* actE = act + (size_t)e * NTOK * IDIM;
  const float*  wd   = w_down + (size_t)e * IDIM * HDIM;

  // Hoisted staging addresses
  const int amb = t >> 4;
  const int akp = t & 15;
  const uint32_t* aRow[8];
  #pragma unroll
  for (int i = 0; i < 8; ++i)
    aRow[i] = (const uint32_t*)(actE + (size_t)(m0 + amb + 16 * i) * IDIM) + akp;
  const int bkb = t >> 6;
  const int bnn = t & 63;
  const float* pd0 = wd + (size_t)bkb * HDIM + n0 + bnn;

  FragC acc[2][2];
  #pragma unroll
  for (int a = 0; a < 2; ++a)
    #pragma unroll
    for (int j = 0; j < 2; ++j)
      #pragma unroll
      for (int r = 0; r < 8; ++r) acc[a][j].f[r] = 0.f;

  for (int k0 = 0; k0 < IDIM; k0 += BK) {
    __syncthreads();
    #pragma unroll
    for (int i = 0; i < 8; ++i)
      *(uint32_t*)&As[amb + 16 * i][akp * 2] = aRow[i][k0 >> 1];
    const float* pd = pd0 + (size_t)k0 * HDIM;
    #pragma unroll
    for (int i = 0; i < 8; ++i)
      Bd[bnn][bkb + 4 * i] = to_bf16(pd[(size_t)(4 * i) * HDIM]);
    if (k0 + BK < IDIM)
      __builtin_prefetch(pd + (size_t)BK * HDIM, 0, 0);
    __syncthreads();

    Frag16 af[2], bd[2];
    load_a_frag(af[0], &As[mB + lr][0], hi);
    load_a_frag(af[1], &As[mB + 16 + lr][0], hi);
    #pragma unroll
    for (int j = 0; j < 2; ++j)
      load_b_frag(bd[j], &Bd[wn * 32 + j * 16 + lr][0], hi);
    #pragma unroll
    for (int a = 0; a < 2; ++a)
      #pragma unroll
      for (int j = 0; j < 2; ++j)
        acc[a][j].v = __builtin_amdgcn_wmma_f32_16x16x32_bf16(
            false, af[a].v, false, bd[j].v, (short)0, acc[a][j].v, false, false);
  }

  #pragma unroll
  for (int a = 0; a < 2; ++a)
    #pragma unroll
    for (int j = 0; j < 2; ++j) {
      const int n = n0 + wn * 32 + j * 16 + lr;
      #pragma unroll
      for (int r = 0; r < 8; ++r) {
        const int ml   = mB + a * 16 + hi * 8 + r;
        const int slot = m0 + ml;
        if (slot < cnt) {
          unsafeAtomicAdd(&out[(size_t)tokS[ml] * HDIM + n], wtS[ml] * acc[a][j].f[r]);
        }
      }
    }
}

// ---------------------------------------------------------------------------
// Launcher. Inputs: hidden_states, gate_w, w_gate, w_up, w_down, top_k, keep_k
// ---------------------------------------------------------------------------
extern "C" void kernel_launch(void* const* d_in, const int* in_sizes, int n_in,
                              void* d_out, int out_size, void* d_ws, size_t ws_size,
                              hipStream_t stream) {
  const float* x      = (const float*)d_in[0];
  const float* gate_w = (const float*)d_in[1];
  const float* w_gate = (const float*)d_in[2];
  const float* w_up   = (const float*)d_in[3];
  const float* w_down = (const float*)d_in[4];
  float* out = (float*)d_out;

  char* ws = (char*)d_ws;
  int*    counts = (int*)ws;                                  // 16 ints (256 B reserved)
  int*    etok   = (int*)(ws + 256);                          // E*N ints
  float*  ewt    = (float*)(ws + 256 + NEXP * NTOK * 4);      // E*N floats
  bf16_t* xb     = (bf16_t*)(ws + 256 + NEXP * NTOK * 8);     // N*H bf16
  bf16_t* act    = xb + (size_t)NTOK * HDIM;                  // E*N*I bf16

  hipMemsetAsync(counts, 0, 256, stream);
  hipMemsetAsync(out, 0, (size_t)NTOK * HDIM * sizeof(float), stream);

  router_kernel<<<NTOK, 256, 0, stream>>>(x, gate_w, xb, counts, etok, ewt);

  gemm1_kernel<<<dim3(IDIM / BN, NTOK / BM, NEXP), 256, 0, stream>>>(
      xb, w_gate, w_up, counts, etok, act);

  gemm2_kernel<<<dim3(HDIM / BN, NTOK / BM, NEXP), 256, 0, stream>>>(
      act, w_down, counts, etok, ewt, out);
}
